// GCN_3255585210653
// MI455X (gfx1250) — matrix operand order repaired and checked by reading, not compiled, not run on hardware
//
#include <hip/hip_runtime.h>
#include <hip/hip_bf16.h>

typedef __attribute__((ext_vector_type(16))) _Float16 v16h;
typedef __attribute__((ext_vector_type(8)))  float    v8f;

#define F_IN 28
#define DEMB 50
#define HID  16
#define NCLS 2

__device__ __forceinline__ void fatomic_add(float* addr, float v) {
    unsafeAtomicAdd(addr, v);   // lowers to global_atomic_add_f32 on gfx1250
}

// ---------------- weight prep: f32 -> f16, zero-padded tiles ----------------
__global__ void prep_params(const float* __restrict__ W1, const float* __restrict__ b1,
                            const float* __restrict__ W2, const float* __restrict__ b2,
                            const float* __restrict__ W3, const float* __restrict__ b3,
                            _Float16* __restrict__ Wp1,   // [64,16]
                            _Float16* __restrict__ Wp2,   // [32,16]
                            _Float16* __restrict__ Wp3,   // [32,16]
                            float* __restrict__ bp)       // [48] = b1|b2|b3pad
{
    int t = threadIdx.x;
    for (int i = t; i < 64 * 16; i += blockDim.x) {
        int k = i >> 4, o = i & 15;
        Wp1[i] = (k < DEMB) ? (_Float16)W1[k * 16 + o] : (_Float16)0.f;
    }
    for (int i = t; i < 32 * 16; i += blockDim.x) {
        int k = i >> 4, o = i & 15;
        Wp2[i] = (k < HID) ? (_Float16)W2[k * 16 + o] : (_Float16)0.f;
        Wp3[i] = (k < HID && o < NCLS) ? (_Float16)W3[k * NCLS + o] : (_Float16)0.f;
    }
    if (t < 16) {
        bp[t]      = b1[t];
        bp[16 + t] = b2[t];
        bp[32 + t] = (t < NCLS) ? b3[t] : 0.f;
    }
}

// ---------------- argmax(x) -> embedding row, padded to 64 f16 cols ---------
__global__ void embed_kernel(const float* __restrict__ x, const float* __restrict__ emb,
                             _Float16* __restrict__ h0, int n)
{
    int nid = blockIdx.x * blockDim.x + threadIdx.x;
    if (nid >= n) return;
    const float* xr = x + (size_t)nid * F_IN;
    int   best = 0;
    float bv   = xr[0];
    #pragma unroll
    for (int j = 1; j < F_IN; ++j) {
        float v = xr[j];
        if (v > bv) { bv = v; best = j; }   // first-max tie break == jnp.argmax
    }
    const float* er = emb + (size_t)best * DEMB;
    _Float16* hr = h0 + (size_t)nid * 64;
    #pragma unroll
    for (int d = 0; d < DEMB; ++d) hr[d] = (_Float16)er[d];
    #pragma unroll
    for (int d = DEMB; d < 64; ++d) hr[d] = (_Float16)0.f;
}

// ---------------- degree / norm prep ----------------------------------------
__global__ void zero_f32(float* __restrict__ p, long total)
{
    long i = (long)blockIdx.x * blockDim.x + threadIdx.x;
    if (i < total) p[i] = 0.f;
}

__global__ void deg_accum(const int* __restrict__ col, const float* __restrict__ ew,
                          float* __restrict__ deg, int e)
{
    int i = blockIdx.x * blockDim.x + threadIdx.x;
    if (i < e) fatomic_add(&deg[col[i]], ew[i]);
}

__global__ void deg_to_dinv(float* __restrict__ d, int n)
{
    int i = blockIdx.x * blockDim.x + threadIdx.x;
    if (i < n) d[i] = rsqrtf(d[i] + 1.0f);   // deg+1 > 0 always
}

__global__ void edge_norm_k(const int* __restrict__ row, const int* __restrict__ col,
                            const float* __restrict__ ew, const float* __restrict__ dinv,
                            float* __restrict__ nrm, int e)
{
    int i = blockIdx.x * blockDim.x + threadIdx.x;
    if (i < e) nrm[i] = dinv[row[i]] * ew[i] * dinv[col[i]];
}

// ---------------- WMMA GEMM: D[N,16] = A[N,K] (f16) x B[K,16] (f16) ---------
// One wave per 16-row tile; K in {32,64}, multiple of 32. N % 16 == 0.
__global__ __launch_bounds__(128) void gemm_wmma(const _Float16* __restrict__ A,
                                                 const _Float16* __restrict__ B,
                                                 float* __restrict__ D,
                                                 int n_rows, int K)
{
    const int wave = threadIdx.x >> 5;
    const int lane = threadIdx.x & 31;
    const int tile = blockIdx.x * 4 + wave;
    const int m0   = tile * 16;
    if (m0 >= n_rows) return;          // wave-uniform: EXEC all-ones below
    const int half = lane >> 4;        // lane group (ISA 16-bit A layout)
    const int m    = lane & 15;
    const _Float16* arow = A + (size_t)(m0 + m) * K;

    v8f c = {};
    for (int kb = 0; kb < K; kb += 32) {
        v16h a, b;
        const int k0 = kb + 8 * half;
        #pragma unroll
        for (int e = 0; e < 8; ++e) a[e] = arow[k0 + e];            // K = kb+8h .. +7
        #pragma unroll
        for (int e = 0; e < 8; ++e) a[8 + e] = arow[k0 + 16 + e];   // K = kb+16+8h .. +7
        const _Float16* brow = B + (size_t)k0 * 16 + m;             // lane&15 = N col of B
        #pragma unroll
        for (int e = 0; e < 8; ++e) { b[e] = brow[e * 16]; b[8 + e] = brow[(16 + e) * 16]; }  // element = K, same order as A
        c = __builtin_amdgcn_wmma_f32_16x16x32_f16(false, a, false, b,
                                                   (short)0, c, false, false);
    }
    // C/D layout: VGPR r -> M = r + 8*half, lane&15 -> N
    float* dcol = D + (size_t)m0 * 16 + m;
    #pragma unroll
    for (int r = 0; r < 8; ++r) dcol[(size_t)(r + 8 * half) * 16] = c[r];
}

// ---------------- self-loop + bias init: agg = hw*dinv^2 + b ----------------
__global__ void agg_init(const float* __restrict__ hw, const float* __restrict__ dinv,
                         const float* __restrict__ bias, float* __restrict__ agg, long total)
{
    long i = (long)blockIdx.x * blockDim.x + threadIdx.x;
    if (i >= total) return;
    int n = (int)(i >> 4), f = (int)(i & 15);
    float di = dinv[n];
    agg[i] = hw[i] * di * di + bias[f];
}

// ---------------- edge scatter: agg[col] += hw[row] * norm ------------------
__global__ void edge_scatter(const int* __restrict__ row, const int* __restrict__ col,
                             const float* __restrict__ nrm, const float* __restrict__ hw,
                             float* __restrict__ agg, long total)
{
    long i = (long)blockIdx.x * blockDim.x + threadIdx.x;
    if (i >= total) return;
    int e = (int)(i >> 4), f = (int)(i & 15);
    float v = hw[(size_t)row[e] * 16 + f] * nrm[e];
    fatomic_add(&agg[(size_t)col[e] * 16 + f], v);
}

// ---------------- ReLU + repack to f16 [N,32] (cols 16..31 zero) ------------
__global__ void relu_to_f16(const float* __restrict__ agg, _Float16* __restrict__ h16, long total)
{
    long i = (long)blockIdx.x * blockDim.x + threadIdx.x;
    if (i >= total) return;
    int n = (int)(i >> 4), f = (int)(i & 15);
    h16[(size_t)n * 32 + f]      = (_Float16)fmaxf(agg[i], 0.f);
    h16[(size_t)n * 32 + 16 + f] = (_Float16)0.f;
}

// ---------------- mean pool + softmax ---------------------------------------
__global__ void pool_accum(const float* __restrict__ h3, const int* __restrict__ batch,
                           float* __restrict__ sums, float* __restrict__ cnt, int n)
{
    int nid = blockIdx.x * blockDim.x + threadIdx.x;
    if (nid >= n) return;
    int g = batch[nid];
    fatomic_add(&sums[(size_t)g * NCLS + 0], h3[(size_t)nid * 16 + 0]);
    fatomic_add(&sums[(size_t)g * NCLS + 1], h3[(size_t)nid * 16 + 1]);
    fatomic_add(&cnt[g], 1.0f);
}

__global__ void softmax_out(const float* __restrict__ sums, const float* __restrict__ cnt,
                            float* __restrict__ out, int g_count)
{
    int g = blockIdx.x * blockDim.x + threadIdx.x;
    if (g >= g_count) return;
    float c  = fmaxf(cnt[g], 1.0f);
    float a  = sums[g * NCLS + 0] / c;
    float b  = sums[g * NCLS + 1] / c;
    float m  = fmaxf(a, b);
    float ea = __expf(a - m), eb = __expf(b - m);
    float s  = ea + eb;
    out[g * NCLS + 0] = ea / s;
    out[g * NCLS + 1] = eb / s;
}

// ============================================================================
extern "C" void kernel_launch(void* const* d_in, const int* in_sizes, int n_in,
                              void* d_out, int out_size, void* d_ws, size_t ws_size,
                              hipStream_t stream)
{
    const float* x     = (const float*)d_in[0];
    const int*   row   = (const int*)  d_in[1];
    const int*   col   = (const int*)  d_in[2];
    const float* ew    = (const float*)d_in[3];
    const int*   batch = (const int*)  d_in[4];
    const float* emb   = (const float*)d_in[5];
    const float* W1    = (const float*)d_in[6];
    const float* b1    = (const float*)d_in[7];
    const float* W2    = (const float*)d_in[8];
    const float* b2    = (const float*)d_in[9];
    const float* W3    = (const float*)d_in[10];
    const float* b3    = (const float*)d_in[11];
    float*       out   = (float*)d_out;

    const int N = in_sizes[0] / F_IN;     // 200000 (divisible by 16)
    const int E = in_sizes[1];            // 3200000
    const int G = out_size / NCLS;        // 8192

    // ---- workspace partition (256B aligned) ----
    char*  ws  = (char*)d_ws;
    size_t off = 0;
    auto carve = [&](size_t bytes) { void* p = ws + off; off = (off + bytes + 255) & ~(size_t)255; return p; };
    _Float16* h0_16  = (_Float16*)carve((size_t)N * 64 * 2);   // layer-1 A (K=64)
    _Float16* h12_16 = (_Float16*)carve((size_t)N * 32 * 2);   // layer-2/3 A (K=32)
    float*    hw     = (float*)   carve((size_t)N * 16 * 4);   // GEMM out
    float*    agg    = (float*)   carve((size_t)N * 16 * 4);   // aggregate
    float*    dinv   = (float*)   carve((size_t)N * 4);        // deg -> d^-1/2
    float*    nrm    = (float*)   carve((size_t)E * 4);        // edge norms
    _Float16* Wp1    = (_Float16*)carve(64 * 16 * 2);
    _Float16* Wp2    = (_Float16*)carve(32 * 16 * 2);
    _Float16* Wp3    = (_Float16*)carve(32 * 16 * 2);
    float*    bp     = (float*)   carve(48 * 4);
    float*    sums   = (float*)   carve((size_t)G * NCLS * 4);
    float*    cnt    = (float*)   carve((size_t)G * 4);
    (void)ws_size; (void)n_in;

    const int  TB = 256;
    const long NF = (long)N * 16;
    const long EF = (long)E * 16;
    auto blocks = [&](long total) { return (unsigned)((total + TB - 1) / TB); };
    const unsigned gemm_blocks = (unsigned)((N / 16 + 3) / 4);

    // ---- preprocessing ----
    prep_params<<<1, TB, 0, stream>>>(W1, b1, W2, b2, W3, b3, Wp1, Wp2, Wp3, bp);
    embed_kernel<<<blocks(N), TB, 0, stream>>>(x, emb, h0_16, N);
    zero_f32<<<blocks(N), TB, 0, stream>>>(dinv, N);
    deg_accum<<<blocks(E), TB, 0, stream>>>(col, ew, dinv, E);
    deg_to_dinv<<<blocks(N), TB, 0, stream>>>(dinv, N);
    edge_norm_k<<<blocks(E), TB, 0, stream>>>(row, col, ew, dinv, nrm, E);

    // ---- layer 1: D=50 (K=64) -> 16, ReLU ----
    gemm_wmma<<<gemm_blocks, 128, 0, stream>>>(h0_16, Wp1, hw, N, 64);
    agg_init<<<blocks(NF), TB, 0, stream>>>(hw, dinv, bp + 0, agg, NF);
    edge_scatter<<<blocks(EF), TB, 0, stream>>>(row, col, nrm, hw, agg, EF);
    relu_to_f16<<<blocks(NF), TB, 0, stream>>>(agg, h12_16, NF);

    // ---- layer 2: 16 (K=32) -> 16, ReLU ----
    gemm_wmma<<<gemm_blocks, 128, 0, stream>>>(h12_16, Wp2, hw, N, 32);
    agg_init<<<blocks(NF), TB, 0, stream>>>(hw, dinv, bp + 16, agg, NF);
    edge_scatter<<<blocks(EF), TB, 0, stream>>>(row, col, nrm, hw, agg, EF);
    relu_to_f16<<<blocks(NF), TB, 0, stream>>>(agg, h12_16, NF);

    // ---- layer 3: 16 (K=32) -> 2 (padded to 16), no activation ----
    gemm_wmma<<<gemm_blocks, 128, 0, stream>>>(h12_16, Wp3, hw, N, 32);
    agg_init<<<blocks(NF), TB, 0, stream>>>(hw, dinv, bp + 32, agg, NF);
    edge_scatter<<<blocks(EF), TB, 0, stream>>>(row, col, nrm, hw, agg, EF);

    // ---- mean pool over graphs + softmax ----
    zero_f32<<<blocks((long)G * NCLS), TB, 0, stream>>>(sums, (long)G * NCLS);
    zero_f32<<<blocks(G), TB, 0, stream>>>(cnt, G);
    pool_accum<<<blocks(N), TB, 0, stream>>>(agg, batch, sums, cnt, N);
    softmax_out<<<blocks(G), TB, 0, stream>>>(sums, cnt, out, G);
}
